// NeuralSDE_forecasting_61134564491966
// MI455X (gfx1250) — matrix operand order repaired and checked by reading, not compiled, not run on hardware
//
#include <hip/hip_runtime.h>
#include <hip/hip_bf16.h>

typedef __attribute__((ext_vector_type(16))) _Float16 v16h;
typedef __attribute__((ext_vector_type(8)))  _Float16 v8h;
typedef __attribute__((ext_vector_type(8)))  float    v8f;

#define HDIM   256
#define CIN    64
#define BDIM   512
#define TDIM   128
#define COUTD  64
#define OUTT   32
#define NSTEP  127
#define HSTART 96      // keep z_{t+1} for t+1 >= T-OUT_T = 96
#define ZLD    264     // padded LDS row stride (halves): 528B -> 4-bank skew, conflict-free b128 reads
#define ZBUF   (16 * ZLD)

static __device__ __forceinline__ v16h cat8(v8h lo, v8h hi) {
  return __builtin_shufflevector(lo, hi, 0,1,2,3,4,5,6,7,8,9,10,11,12,13,14,15);
}

static __device__ __forceinline__ v8f wmma16(v16h a, v16h b, v8f c) {
  // emits v_wmma_f32_16x16x32_f16
  return __builtin_amdgcn_wmma_f32_16x16x32_f16(false, a, false, b, (short)0, c, false, false);
}

// A-fragment (16x32 f16): lane<16 -> row=lane, K = kb+{0..7} then kb+{16..23};
// lane>=16 -> row=lane-16, K = kb+{8..15} then kb+{24..31}. (ISA 7.12.2)
static __device__ __forceinline__ v16h frag_a(const _Float16* mat, int ld, int row_base, int k_base, int lane) {
  int r = lane & 15, hl = lane >> 4;
  const _Float16* p = mat + (row_base + r) * ld + k_base + hl * 8;
  v8h lo = *(const v8h*)p;
  v8h hi = *(const v8h*)(p + 16);
  return cat8(lo, hi);
}

// B-fragment (32x16 f16) from a column-major (N-major, K-contiguous) matrix:
// lane<16 -> N=nb+lane, K=kb+{0..7} then kb+{16..23} ; lane>=16 -> N=nb+lane-16, K=kb+{8..15} then kb+{24..31}.
static __device__ __forceinline__ v16h frag_b(const _Float16* wt, int ld, int n_base, int k_base, int lane) {
  int r = lane & 15, hl = lane >> 4;
  const _Float16* p = wt + (n_base + r) * ld + k_base + hl * 8;
  v8h lo = *(const v8h*)p;
  v8h hi = *(const v8h*)(p + 16);
  return cat8(lo, hi);
}

// ---------------- prep: dt, f16 transposed weights, x0 -> f16 ----------------
__global__ __launch_bounds__(256) void sde_prep(
    const float* __restrict__ times, const float* __restrict__ coeffs,
    const float* __restrict__ W_init, const float* __restrict__ Wf,
    const float* __restrict__ Wg, const float* __restrict__ W1,
    const float* __restrict__ W2, float* __restrict__ dts,
    _Float16* __restrict__ WfT, _Float16* __restrict__ WgT,
    _Float16* __restrict__ W1T, _Float16* __restrict__ W2T,
    _Float16* __restrict__ WiT, _Float16* __restrict__ x0h) {
  int gid = blockIdx.x * blockDim.x + threadIdx.x;
  int gsz = gridDim.x * blockDim.x;
  if (gid == 0) {
    float m = 3.4e38f;
    for (int i = 1; i < TDIM; ++i) { float d = times[i] - times[i - 1]; m = d < m ? d : m; }
    float dt = m > 0.001f ? m : 0.001f;
    dts[0] = dt; dts[1] = sqrtf(dt);
  }
  for (int i = gid; i < HDIM * HDIM; i += gsz) {           // [n*256+k] = W[k,n]
    int n = i >> 8, k = i & 255;
    WfT[i] = (_Float16)Wf[k * HDIM + n];
    WgT[i] = (_Float16)Wg[k * HDIM + n];
    W1T[i] = (_Float16)W1[k * HDIM + n];
  }
  for (int i = gid; i < COUTD * HDIM; i += gsz) {          // W2: (256,64) -> [n*256+k]
    int n = i >> 8, k = i & 255;
    W2T[i] = (_Float16)W2[k * COUTD + n];
  }
  for (int i = gid; i < HDIM * CIN; i += gsz) {            // W_init: (64,256) -> [n*64+k]
    int n = i >> 6, k = i & 63;
    WiT[i] = (_Float16)W_init[k * HDIM + n];
  }
  for (int i = gid; i < BDIM * CIN; i += gsz) {            // x0 = coeffs[:,0,:]
    int b = i >> 6, c = i & 63;
    x0h[i] = (_Float16)coeffs[(size_t)b * TDIM * CIN + c];
  }
}

// ---- scan: 32 WGs x 16 waves; weights in registers; double-buffered z tile ----
__global__ __launch_bounds__(512) void sde_scan(
    const _Float16* __restrict__ x0h, const _Float16* __restrict__ WiT,
    const _Float16* __restrict__ WfT, const _Float16* __restrict__ WgT,
    const float* __restrict__ b_init, const float* __restrict__ bfb,
    const float* __restrict__ bgb, const float* __restrict__ noise,
    const float* __restrict__ dts, _Float16* __restrict__ hbuf) {
  __shared__ __align__(16) _Float16 zs[2 * ZBUF];   // ping-pong 16-row z tiles, f16
  int tid = threadIdx.x, wave = tid >> 5, lane = tid & 31;
  int b0 = blockIdx.x * 16;           // 16 batch rows per WG
  int n0 = wave * 16;                 // each wave owns one 16-col N-tile
  int r15 = lane & 15, hl = lane >> 4;
  int col = n0 + r15;
  float dt = dts[0], sq = dts[1];
  float bfv = bfb[col], bgv = bgb[col];

  // preload loop-invariant weight B-fragments into registers (2 x 8 x v16h = 128 VGPRs)
  v16h wfr[8], wgr[8];
#pragma unroll
  for (int kt = 0; kt < 8; ++kt) {
    wfr[kt] = frag_b(WfT, HDIM, n0, kt * 32, lane);
    wgr[kt] = frag_b(WgT, HDIM, n0, kt * 32, lane);
  }

  // stage x0 tile (16x64) into buffer 1; loop iteration 0 writes buffer 0,
  // so the z0 fragment reads below are ordered by barrier(t=0) alone.
  for (int i = tid; i < 16 * CIN; i += 512) {
    int r = i >> 6, c = i & 63;
    zs[ZBUF + r * ZLD + c] = x0h[(b0 + r) * CIN + c];
  }
  __syncthreads();

  v8f z;
  {
    v16h a0 = frag_a(zs + ZBUF, ZLD, 0, 0, lane);
    v16h a1 = frag_a(zs + ZBUF, ZLD, 0, 32, lane);
    v8f acc = {};
    acc = wmma16(a0, frag_b(WiT, CIN, n0, 0, lane), acc);
    acc = wmma16(a1, frag_b(WiT, CIN, n0, 32, lane), acc);
    float bv = b_init[col];
#pragma unroll
    for (int e = 0; e < 8; ++e) acc[e] += bv;
    z = acc;
  }

  for (int t = 0; t < NSTEP; ++t) {
    _Float16* buf = zs + (t & 1) * ZBUF;
    // write wave's z slice (16x16) into this step's buffer (C-layout -> row-major).
    // Safe without a pre-barrier: the previous writer of this buffer was step t-2,
    // and its readers finished before barrier(t-1) < this point.
#pragma unroll
    for (int e = 0; e < 8; ++e)
      buf[(e + 8 * hl) * ZLD + col] = (_Float16)z[e];
    __syncthreads();                  // single barrier per step: writes -> reads

    v16h a[8];
#pragma unroll
    for (int kt = 0; kt < 8; ++kt) a[kt] = frag_a(buf, ZLD, 0, kt * 32, lane);

    // prefetch next step's noise while the WMMA chain runs
    if (t + 1 < NSTEP)
      __builtin_prefetch(&noise[((size_t)(t + 1) * BDIM + b0 + 8 * hl) * HDIM + col], 0, 1);

    v8f fa = {}, ga = {};
#pragma unroll
    for (int kt = 0; kt < 8; ++kt) {
      fa = wmma16(a[kt], wfr[kt], fa);
      ga = wmma16(a[kt], wgr[kt], ga);
    }

    int j = t + 1 - HSTART;
#pragma unroll
    for (int e = 0; e < 8; ++e) {
      int rowg = b0 + e + 8 * hl;
      float pf = fa[e] + bfv;
      float ex = __expf(-2.0f * pf);
      float fv = (1.0f - ex) / (1.0f + ex);            // tanh
      float pg = ga[e] + bgv;
      float gv = 1.0f / (1.0f + __expf(-pg));          // sigmoid
      float eps = noise[((size_t)t * BDIM + rowg) * HDIM + col];
      float zn = z[e] + fv * dt + gv * sq * eps;
      z[e] = zn;
      if (j >= 0)
        hbuf[((size_t)rowg * OUTT + j) * HDIM + col] = (_Float16)zn;
    }
  }
}

// ---------------- head: relu(h@W1+b1)@W2+b2, fused through LDS ----------------
__global__ __launch_bounds__(256) void sde_head(
    const _Float16* __restrict__ hbuf, const _Float16* __restrict__ W1T,
    const _Float16* __restrict__ W2T, const float* __restrict__ b1,
    const float* __restrict__ b2, float* __restrict__ out) {
  __shared__ __align__(16) _Float16 y1[64 * ZLD];
  int tid = threadIdx.x, wave = tid >> 5, lane = tid & 31;
  int r15 = lane & 15, hl = lane >> 4;
  int R0 = blockIdx.x * 64;           // 64 rows of (16384 x 256) per WG
  int lr = (wave & 3) * 16;           // wave M-tile
  v16h a[8];
#pragma unroll
  for (int kt = 0; kt < 8; ++kt) a[kt] = frag_a(hbuf, HDIM, R0 + lr, kt * 32, lane);
  int ntb = (wave >> 2) * 8;          // wave N-half of stage-1
  for (int nt = ntb; nt < ntb + 8; ++nt) {
    int nb = nt * 16;
    v8f acc = {};
#pragma unroll
    for (int kt = 0; kt < 8; ++kt)
      acc = wmma16(a[kt], frag_b(W1T, HDIM, nb, kt * 32, lane), acc);
    float bv = b1[nb + r15];
#pragma unroll
    for (int e = 0; e < 8; ++e) {
      float v = acc[e] + bv;
      v = v > 0.f ? v : 0.f;                             // relu
      y1[(lr + e + 8 * hl) * ZLD + nb + r15] = (_Float16)v;
    }
  }
  __syncthreads();
#pragma unroll
  for (int kt = 0; kt < 8; ++kt) a[kt] = frag_a(y1, ZLD, lr, kt * 32, lane);
  int ntp = (wave >> 2) * 2;          // COUT = 4 n-tiles, 2 per wave
  for (int nt = ntp; nt < ntp + 2; ++nt) {
    int nb = nt * 16;
    v8f acc = {};
#pragma unroll
    for (int kt = 0; kt < 8; ++kt)
      acc = wmma16(a[kt], frag_b(W2T, HDIM, nb, kt * 32, lane), acc);
    float bv = b2[nb + r15];
#pragma unroll
    for (int e = 0; e < 8; ++e)
      out[(size_t)(R0 + lr + e + 8 * hl) * COUTD + nb + r15] = acc[e] + bv;
  }
}

extern "C" void kernel_launch(void* const* d_in, const int* in_sizes, int n_in,
                              void* d_out, int out_size, void* d_ws, size_t ws_size,
                              hipStream_t stream) {
  const float* times  = (const float*)d_in[0];
  const float* coeffs = (const float*)d_in[1];
  const float* noise  = (const float*)d_in[2];
  const float* W_init = (const float*)d_in[3];
  const float* b_init = (const float*)d_in[4];
  const float* Wf     = (const float*)d_in[5];
  const float* bf     = (const float*)d_in[6];
  const float* Wg     = (const float*)d_in[7];
  const float* bg     = (const float*)d_in[8];
  const float* W1     = (const float*)d_in[9];
  const float* b1     = (const float*)d_in[10];
  const float* W2     = (const float*)d_in[11];
  const float* b2     = (const float*)d_in[12];
  (void)in_sizes; (void)n_in; (void)out_size; (void)ws_size;

  char* ws = (char*)d_ws;
  float*    dts  = (float*)(ws + 0);
  _Float16* WfT  = (_Float16*)(ws + 64);
  _Float16* WgT  = (_Float16*)(ws + 64 + 131072);
  _Float16* W1T  = (_Float16*)(ws + 64 + 2 * 131072);
  _Float16* W2T  = (_Float16*)(ws + 64 + 3 * 131072);
  _Float16* WiT  = (_Float16*)(ws + 64 + 3 * 131072 + 32768);
  _Float16* x0h  = (_Float16*)(ws + 64 + 3 * 131072 + 2 * 32768);
  _Float16* hbuf = (_Float16*)(ws + 64 + 3 * 131072 + 2 * 32768 + 65536);

  sde_prep<<<256, 256, 0, stream>>>(times, coeffs, W_init, Wf, Wg, W1, W2,
                                    dts, WfT, WgT, W1T, W2T, WiT, x0h);
  sde_scan<<<32, 512, 0, stream>>>(x0h, WiT, WfT, WgT, b_init, bf, bg, noise, dts, hbuf);
  sde_head<<<256, 256, 0, stream>>>(hbuf, W1T, W2T, b1, b2, (float*)d_out);
}
